// LSTM_autoreg_torchscript_physrad_60567628808894
// MI455X (gfx1250) — matrix operand order repaired and checked by reading, not compiled, not run on hardware
//
#include <hip/hip_runtime.h>
#include <hip/hip_bf16.h>

// ---------------- problem constants ----------------
#define NLEV   60
#define NXF    15
#define NXM_   12
#define NMEM_  16
#define NXS_   24
#define NXSR   19
#define NTOA_  2
#define NH     192     // NH1 == NH2
#define NG     768     // 4*NH gate width
#define XMK    32      // padded input-feature K (28 -> 32)
#define BM     16      // batch rows per workgroup (one WMMA M-tile)
#define NWAVES 12      // one wave per 16-col j-tile of NH=192
#define THREADS (NWAVES * 32)

typedef __attribute__((ext_vector_type(16))) _Float16     v16h;
typedef __attribute__((ext_vector_type(8)))  float        v8f;
typedef __attribute__((ext_vector_type(4)))  unsigned int u32x4;

union FragU { v16h v; u32x4 q[2]; };

// A fragment (16Mx32K f16): lane l<16 -> row l%16, K {kb+0..7, kb+16..23};
// lane>=16 -> same row, K {kb+8..15, kb+24..31}.  p points at K=kb of the row.
__device__ __forceinline__ v16h load_frag_a(const _Float16* p, int klo) {
  FragU f;
  f.q[0] = *(const u32x4*)(p + klo);
  f.q[1] = *(const u32x4*)(p + 16 + klo);
  return f.v;
}
// B fragment (32Kx16N f16): lane l<16 -> col l, K {kb+0..7, kb+16..23}; lane>=16 -> col l-16,
// K {kb+8..15, kb+24..31}.  p points at K=kb of the lane's column (row-major W[n][k]).
__device__ __forceinline__ v16h load_frag_b(const _Float16* p, int koff) {
  FragU f;
  f.q[0] = *(const u32x4*)(p + koff);
  f.q[1] = *(const u32x4*)(p + 16 + koff);
  return f.v;
}

__device__ __forceinline__ float sigm(float x) { return 1.0f / (1.0f + __expf(-x)); }

// Branch-free tanh: native v_tanh_f32 if available, else exp-based form that
// saturates correctly at +-1 (exp overflow -> +1, underflow -> -1).
__device__ __forceinline__ float ftanh(float x) {
#if __has_builtin(__builtin_amdgcn_tanhf)
  return __builtin_amdgcn_tanhf(x);
#else
  return 1.0f - 2.0f / (1.0f + __expf(2.0f * x));
#endif
}

// ---------------- init kernels ----------------

// xm[b,t,i] = normalized/flipped/padded input features, fp16, K padded to 32.
__global__ void k_build_xm(const float* __restrict__ x_lev,
                           const float* __restrict__ memory,
                           const float* __restrict__ xmean,
                           const float* __restrict__ xdiv,
                           _Float16* __restrict__ xm, int B) {
  int idx = blockIdx.x * blockDim.x + threadIdx.x;
  int total = B * NLEV * XMK;
  if (idx >= total) return;
  int i  = idx & (XMK - 1);
  int bt = idx >> 5;
  int t  = bt % NLEV;
  int b  = bt / NLEV;
  int l  = NLEV - 1 - t;                       // flip fused here
  float v = 0.0f;
  if (i < NXM_)       v = (x_lev[(b * NLEV + l) * NXF + i] - xmean[l * NXF + i]) / xdiv[l * NXF + i];
  else if (i < NXM_ + NMEM_) v = memory[(b * NLEV + l) * NMEM_ + (i - NXM_)];
  xm[idx] = (_Float16)v;
}

__global__ void k_cvt(const float* __restrict__ s, _Float16* __restrict__ d, int n) {
  int i = blockIdx.x * blockDim.x + threadIdx.x;
  if (i < n) d[i] = (_Float16)s[i];
}

// W_ih1 (768,28) -> fp16 (768,32) zero padded
__global__ void k_cvt_pad(const float* __restrict__ s, _Float16* __restrict__ d, int n) {
  int idx = blockIdx.x * blockDim.x + threadIdx.x;
  if (idx >= n) return;
  int i = idx & (XMK - 1);
  int r = idx >> 5;
  d[idx] = (i < 28) ? (_Float16)s[r * 28 + i] : (_Float16)0.0f;
}

// h0/c0 (both LSTMs): tiny K=19 / K=2 dots + tanh
__global__ void k_init_state(const float* __restrict__ x_sfc,
                             const float* __restrict__ xmean_sca,
                             const float* __restrict__ xdiv_sca,
                             const float* __restrict__ W_sfc1, const float* __restrict__ b_sfc1,
                             const float* __restrict__ W_sfc2, const float* __restrict__ b_sfc2,
                             const float* __restrict__ W_toa1, const float* __restrict__ b_toa1,
                             const float* __restrict__ W_toa2, const float* __restrict__ b_toa2,
                             _Float16* __restrict__ h0a, float* __restrict__ c0a,
                             _Float16* __restrict__ h0b, float* __restrict__ c0b, int B) {
  int idx = blockIdx.x * blockDim.x + threadIdx.x;
  int per = B * NH;
  if (idx >= 4 * per) return;
  int which = idx / per;
  int rem   = idx % per;
  int b = rem / NH, h = rem % NH;
  const float* W; const float* bb; int K;
  if (which == 0)      { W = W_sfc1; bb = b_sfc1; K = NXSR; }
  else if (which == 1) { W = W_sfc2; bb = b_sfc2; K = NXSR; }
  else if (which == 2) { W = W_toa1; bb = b_toa1; K = NTOA_; }
  else                 { W = W_toa2; bb = b_toa2; K = NTOA_; }
  float s = bb[h];
  for (int k = 0; k < K; ++k) {
    float xs = (x_sfc[b * NXS_ + k] - xmean_sca[k]) / xdiv_sca[k];
    s += xs * W[h * K + k];
  }
  s = ftanh(s);
  if (which == 0)      h0a[rem] = (_Float16)s;
  else if (which == 1) c0a[rem] = s;
  else if (which == 2) h0b[rem] = (_Float16)s;
  else                 c0b[rem] = s;
}

// ---------------- LSTM layer 1 (reversed sequence) ----------------
__global__ __launch_bounds__(THREADS) void k_lstm1(
    const _Float16* __restrict__ xm,    // (B,60,32) fp16, flipped+padded
    const _Float16* __restrict__ wih,   // (768,32)  fp16 padded
    const _Float16* __restrict__ whh,   // (768,192) fp16
    const float* __restrict__ b1,       // (768)
    const _Float16* __restrict__ h0,    // (B,192) fp16
    const float* __restrict__ c0,       // (B,192) f32
    _Float16* __restrict__ y1)          // (B,60,192) fp16 (stored at 59-t)
{
  __shared__ __align__(16) _Float16 h_lds[BM * NH];
  const int tid  = threadIdx.x;
  const int lane = tid & 31;
  const int jt   = tid >> 5;             // wave id == j-tile (0..11)
  const int ln   = lane & 15;
  const int hi16 = (lane >= 16);
  const int rowbase = blockIdx.x * BM;
  const int j    = jt * 16 + ln;         // hidden column (WMMA N & C-layout col)
  const int kloA = hi16 ? 8 : 0;
  const int koffB = hi16 ? 8 : 0;

  for (int i = tid; i < BM * NH; i += THREADS)
    h_lds[i] = h0[(size_t)(rowbase + i / NH) * NH + (i % NH)];

  float bv[4];
  v8f c;
#pragma unroll
  for (int g = 0; g < 4; ++g) bv[g] = b1[g * NH + j];
#pragma unroll
  for (int r = 0; r < 8; ++r)
    c[r] = c0[(size_t)(rowbase + r + (hi16 ? 8 : 0)) * NH + j];
  __syncthreads();

  const _Float16* xrow = xm + ((size_t)(rowbase + ln) * NLEV) * XMK;

  for (int t = 0; t < NLEV; ++t) {
    v8f acc[4];
#pragma unroll
    for (int g = 0; g < 4; ++g)
#pragma unroll
      for (int r = 0; r < 8; ++r) acc[g][r] = bv[g];

    // input GEMM: one K=32 step (features padded 28->32)
    {
      v16h a = load_frag_a(xrow + (size_t)t * XMK, kloA);
#pragma unroll
      for (int g = 0; g < 4; ++g) {
        v16h b = load_frag_b(wih + (size_t)(g * NH + j) * XMK, koffB);
        acc[g] = __builtin_amdgcn_wmma_f32_16x16x32_f16(false, a, false, b, (short)0, acc[g], false, false);
      }
    }
    // recurrent GEMM: K = 192 in 6 steps, A from LDS h, B weights read from global each step
#pragma unroll
    for (int kb = 0; kb < NH; kb += 32) {
      v16h a = load_frag_a(&h_lds[ln * NH + kb], kloA);
#pragma unroll
      for (int g = 0; g < 4; ++g) {
        v16h b = load_frag_b(whh + (size_t)(g * NH + j) * NH + kb, koffB);
        acc[g] = __builtin_amdgcn_wmma_f32_16x16x32_f16(false, a, false, b, (short)0, acc[g], false, false);
      }
    }
    // gate fusion: lane-local thanks to matching C layouts of all 4 gates
    _Float16 hh[8];
#pragma unroll
    for (int r = 0; r < 8; ++r) {
      float ig = sigm(acc[0][r]);
      float fg = sigm(acc[1][r]);
      float gg = ftanh(acc[2][r]);
      float og = sigm(acc[3][r]);
      float cn = fg * c[r] + ig * gg;
      c[r] = cn;
      hh[r] = (_Float16)(og * ftanh(cn));
    }
    __syncthreads();                       // all waves done reading old h
    const int lout = NLEV - 1 - t;         // un-flip on store
#pragma unroll
    for (int r = 0; r < 8; ++r) {
      int m = r + (hi16 ? 8 : 0);
      h_lds[m * NH + j] = hh[r];
      y1[((size_t)(rowbase + m) * NLEV + lout) * NH + j] = hh[r];
    }
    __syncthreads();                       // new h visible
  }
}

// ---------------- LSTM layer 2 + fused output heads ----------------
__global__ __launch_bounds__(THREADS) void k_lstm2(
    const _Float16* __restrict__ y1,    // (B,60,192) fp16
    const _Float16* __restrict__ wih,   // (768,192) fp16
    const _Float16* __restrict__ whh,   // (768,192) fp16
    const float* __restrict__ b2,
    const _Float16* __restrict__ h0,
    const float* __restrict__ c0,
    const float* __restrict__ W_lat, const float* __restrict__ b_lat,
    const float* __restrict__ W_out, const float* __restrict__ b_out,
    const float* __restrict__ W_sfco, const float* __restrict__ b_sfco,
    const float* __restrict__ ysl,      // (60,5)
    const float* __restrict__ yss,      // (2)
    float* __restrict__ out, int B)
{
  __shared__ __align__(16) _Float16 h_lds[BM * NH];
  __shared__ float lat_lds[BM * 16];
  const int tid  = threadIdx.x;
  const int lane = tid & 31;
  const int jt   = tid >> 5;
  const int ln   = lane & 15;
  const int hi16 = (lane >= 16);
  const int rowbase = blockIdx.x * BM;
  const int j    = jt * 16 + ln;
  const int kloA = hi16 ? 8 : 0;
  const int koffB = hi16 ? 8 : 0;

  for (int i = tid; i < BM * NH; i += THREADS)
    h_lds[i] = h0[(size_t)(rowbase + i / NH) * NH + (i % NH)];

  float bv[4];
  v8f c;
#pragma unroll
  for (int g = 0; g < 4; ++g) bv[g] = b2[g * NH + j];
#pragma unroll
  for (int r = 0; r < 8; ++r)
    c[r] = c0[(size_t)(rowbase + r + (hi16 ? 8 : 0)) * NH + j];
  __syncthreads();

  const _Float16* yrow = y1 + ((size_t)(rowbase + ln) * NLEV) * NH;

  for (int t = 0; t < NLEV; ++t) {
    v8f acc[4];
#pragma unroll
    for (int g = 0; g < 4; ++g)
#pragma unroll
      for (int r = 0; r < 8; ++r) acc[g][r] = bv[g];

    // input GEMM: xg2 = y1[:,t] @ W_ih2^T, K = 192
#pragma unroll
    for (int kb = 0; kb < NH; kb += 32) {
      v16h a = load_frag_a(yrow + (size_t)t * NH + kb, kloA);
#pragma unroll
      for (int g = 0; g < 4; ++g) {
        v16h b = load_frag_b(wih + (size_t)(g * NH + j) * NH + kb, koffB);
        acc[g] = __builtin_amdgcn_wmma_f32_16x16x32_f16(false, a, false, b, (short)0, acc[g], false, false);
      }
    }
    // recurrent GEMM
#pragma unroll
    for (int kb = 0; kb < NH; kb += 32) {
      v16h a = load_frag_a(&h_lds[ln * NH + kb], kloA);
#pragma unroll
      for (int g = 0; g < 4; ++g) {
        v16h b = load_frag_b(whh + (size_t)(g * NH + j) * NH + kb, koffB);
        acc[g] = __builtin_amdgcn_wmma_f32_16x16x32_f16(false, a, false, b, (short)0, acc[g], false, false);
      }
    }
    _Float16 hh[8];
#pragma unroll
    for (int r = 0; r < 8; ++r) {
      float ig = sigm(acc[0][r]);
      float fg = sigm(acc[1][r]);
      float gg = ftanh(acc[2][r]);
      float og = sigm(acc[3][r]);
      float cn = fg * c[r] + ig * gg;
      c[r] = cn;
      hh[r] = (_Float16)(og * ftanh(cn));
    }
    __syncthreads();
#pragma unroll
    for (int r = 0; r < 8; ++r)
      h_lds[(r + (hi16 ? 8 : 0)) * NH + j] = hh[r];
    __syncthreads();                        // h_t (== y2[:,t]) in LDS

    // fused head: lat = y2 @ W_lat^T + b_lat  (16 rows x 16 cols, K=192)
    for (int task = tid; task < BM * 16; task += THREADS) {
      int m = task >> 4, jl = task & 15;
      float s = b_lat[jl];
      for (int k = 0; k < NH; ++k)
        s += (float)h_lds[m * NH + k] * W_lat[jl * NH + k];
      lat_lds[m * 16 + jl] = s;
    }
    __syncthreads();
    // out = (lat @ W_out^T + b_out) / yscale_lev[t]
    for (int task = tid; task < BM * 5; task += THREADS) {
      int m = task / 5, y = task % 5;
      float s = b_out[y];
      for (int k = 0; k < 16; ++k) s += lat_lds[m * 16 + k] * W_out[y * 16 + k];
      out[(size_t)(rowbase + m) * (NLEV * 5) + t * 5 + y] = s / ysl[t * 5 + y];
    }
    if (t == NLEV - 1) {                    // out_sfc from h2_last
      for (int task = tid; task < BM * 2; task += THREADS) {
        int m = task >> 1, s2 = task & 1;
        float s = b_sfco[s2];
        for (int k = 0; k < NH; ++k)
          s += (float)h_lds[m * NH + k] * W_sfco[s2 * NH + k];
        out[(size_t)B * (NLEV * 5) + (size_t)(rowbase + m) * 2 + s2] = s / yss[s2];
      }
    }
  }
}

// ---------------- launcher ----------------
extern "C" void kernel_launch(void* const* d_in, const int* in_sizes, int n_in,
                              void* d_out, int out_size, void* d_ws, size_t ws_size,
                              hipStream_t stream) {
  const float* x_lev     = (const float*)d_in[0];
  const float* x_sfc     = (const float*)d_in[1];
  const float* memory    = (const float*)d_in[2];
  const float* xmean_lev = (const float*)d_in[3];
  const float* xdiv_lev  = (const float*)d_in[4];
  const float* xmean_sca = (const float*)d_in[5];
  const float* xdiv_sca  = (const float*)d_in[6];
  const float* yscale_lev= (const float*)d_in[7];
  const float* yscale_sca= (const float*)d_in[8];
  const float* W_sfc1 = (const float*)d_in[9];
  const float* b_sfc1 = (const float*)d_in[10];
  const float* W_sfc2 = (const float*)d_in[11];
  const float* b_sfc2 = (const float*)d_in[12];
  const float* W_toa1 = (const float*)d_in[13];
  const float* b_toa1 = (const float*)d_in[14];
  const float* W_toa2 = (const float*)d_in[15];
  const float* b_toa2 = (const float*)d_in[16];
  const float* W_ih1  = (const float*)d_in[17];
  const float* W_hh1  = (const float*)d_in[18];
  const float* b_1    = (const float*)d_in[19];
  const float* W_ih2  = (const float*)d_in[20];
  const float* W_hh2  = (const float*)d_in[21];
  const float* b_2    = (const float*)d_in[22];
  const float* W_lat  = (const float*)d_in[23];
  const float* b_lat  = (const float*)d_in[24];
  const float* W_out  = (const float*)d_in[25];
  const float* b_out  = (const float*)d_in[26];
  const float* W_sfco = (const float*)d_in[27];
  const float* b_sfco = (const float*)d_in[28];
  const int B = in_sizes[0] / (NLEV * NXF);

  char* w = (char*)d_ws;
  size_t off = 0;
  auto alloc = [&](size_t bytes) {
    void* p = w + off;
    off += (bytes + 255) & ~(size_t)255;
    return p;
  };
  _Float16* xm    = (_Float16*)alloc((size_t)B * NLEV * XMK * sizeof(_Float16));
  _Float16* wih1h = (_Float16*)alloc((size_t)NG * XMK * sizeof(_Float16));
  _Float16* whh1h = (_Float16*)alloc((size_t)NG * NH * sizeof(_Float16));
  _Float16* wih2h = (_Float16*)alloc((size_t)NG * NH * sizeof(_Float16));
  _Float16* whh2h = (_Float16*)alloc((size_t)NG * NH * sizeof(_Float16));
  _Float16* h0a   = (_Float16*)alloc((size_t)B * NH * sizeof(_Float16));
  float*    c0a   = (float*)   alloc((size_t)B * NH * sizeof(float));
  _Float16* h0b   = (_Float16*)alloc((size_t)B * NH * sizeof(_Float16));
  float*    c0b   = (float*)   alloc((size_t)B * NH * sizeof(float));
  _Float16* y1    = (_Float16*)alloc((size_t)B * NLEV * NH * sizeof(_Float16));

  int n;
  n = B * NLEV * XMK;
  k_build_xm<<<(n + 255) / 256, 256, 0, stream>>>(x_lev, memory, xmean_lev, xdiv_lev, xm, B);
  n = NG * XMK;
  k_cvt_pad<<<(n + 255) / 256, 256, 0, stream>>>(W_ih1, wih1h, n);
  n = NG * NH;
  k_cvt<<<(n + 255) / 256, 256, 0, stream>>>(W_hh1, whh1h, n);
  k_cvt<<<(n + 255) / 256, 256, 0, stream>>>(W_ih2, wih2h, n);
  k_cvt<<<(n + 255) / 256, 256, 0, stream>>>(W_hh2, whh2h, n);
  n = 4 * B * NH;
  k_init_state<<<(n + 255) / 256, 256, 0, stream>>>(
      x_sfc, xmean_sca, xdiv_sca, W_sfc1, b_sfc1, W_sfc2, b_sfc2,
      W_toa1, b_toa1, W_toa2, b_toa2, h0a, c0a, h0b, c0b, B);

  k_lstm1<<<B / BM, THREADS, 0, stream>>>(xm, wih1h, whh1h, b_1, h0a, c0a, y1);
  k_lstm2<<<B / BM, THREADS, 0, stream>>>(y1, wih2h, whh2h, b_2, h0b, c0b,
                                          W_lat, b_lat, W_out, b_out, W_sfco, b_sfco,
                                          yscale_lev, yscale_sca, (float*)d_out, B);
}
